// ContrastiveLoss_58110907514905
// MI455X (gfx1250) — matrix operand order repaired and checked by reading, not compiled, not run on hardware
//
#include <hip/hip_runtime.h>

// ---------------------------------------------------------------------------
// NT-Xent contrastive loss for MI455X (gfx1250, wave32, WMMA).
// sim = (F F^T)/T with F = rows of [normalize(f1); normalize(f2)], f16.
//
// v3: B tiles staged into LDS with CDNA5 GLOBAL_LOAD_ASYNC_TO_LDS_B128
// (ASYNCcnt-tracked DMA, inline asm) so the copy overlaps the 8-WMMA chain;
// the only wait is s_wait_asynccnt 0 immediately before the publishing
// barrier. Grid = 64 row-strips (128 rows) x 4 column-quarters.
// ---------------------------------------------------------------------------

typedef __attribute__((ext_vector_type(16))) _Float16 v16h;
typedef __attribute__((ext_vector_type(8)))  _Float16 v8h;
typedef __attribute__((ext_vector_type(8)))  float    v8f;
typedef __attribute__((ext_vector_type(4)))  float    v4f;

#define NROWS    8192            // 2B
#define BHALF    4096            // B
#define DIM      256             // D
#define TILE     16
#define KCH      8               // DIM / 32 (K-chunks per WMMA chain)
#define WAVES    8               // waves per workgroup
#define STRIP    128             // rows per workgroup (WAVES * TILE)
#define NSTRIPS  (NROWS / STRIP) // 64
#define NQ       4               // column quarters
#define QCOLS    (NROWS / NQ)    // 2048
#define QTILES   (QCOLS / TILE)  // 128
#define LDSPITCH 264             // halves per B-tile row in LDS (528B = 132dw bank skew)

static constexpr float INV_T = 1.0f / 0.07f;   // logit bound -> exp shift

// --- CDNA5 async copy helpers (inline asm; ASYNCcnt-tracked) ----------------
typedef __attribute__((address_space(3))) const void lds_cvoid_t;

static __device__ __forceinline__ uint32_t lds_byte_addr(const void* p) {
  // generic -> LDS addrspace cast, then to the 32-bit LDS byte offset
  return (uint32_t)(uintptr_t)(lds_cvoid_t*)p;
}

static __device__ __forceinline__ void async_copy_b128(uint32_t lds_byte,
                                                       const void* gptr) {
  asm volatile("global_load_async_to_lds_b128 %0, %1, off"
               :: "v"(lds_byte), "v"((uint64_t)(uintptr_t)gptr)
               : "memory");
}

static __device__ __forceinline__ void wait_asynccnt0() {
  asm volatile("s_wait_asynccnt 0" ::: "memory");
}

// ---------------------------------------------------------------------------
// Kernel 1: L2-normalize rows of both views; write f16 F[8192][256] to ws.
// One wave per row: 32 lanes x 8 floats = 256.
// ---------------------------------------------------------------------------
__global__ __launch_bounds__(256) void ContrastiveLoss_nrm(
    const float* __restrict__ f1, const float* __restrict__ f2,
    _Float16* __restrict__ F) {
  const int wave = threadIdx.x >> 5;
  const int lane = threadIdx.x & 31;
  const int row  = blockIdx.x * 8 + wave;

  const float* src = (row < BHALF) ? (f1 + (size_t)row * DIM)
                                   : (f2 + (size_t)(row - BHALF) * DIM);
  v4f x0 = *(const v4f*)(src + lane * 8);
  v4f x1 = *(const v4f*)(src + lane * 8 + 4);

  float ss = 0.0f;
#pragma unroll
  for (int i = 0; i < 4; ++i) { ss += x0[i] * x0[i]; ss += x1[i] * x1[i]; }
#pragma unroll
  for (int m = 16; m >= 1; m >>= 1) ss += __shfl_xor(ss, m, 32);

  const float scale = 1.0f / fmaxf(sqrtf(ss), 1e-12f);

  v8h out;
#pragma unroll
  for (int i = 0; i < 4; ++i) {
    out[i]     = (_Float16)(x0[i] * scale);
    out[4 + i] = (_Float16)(x1[i] * scale);
  }
  *(v8h*)(F + (size_t)row * DIM + lane * 8) = out;
}

// ---------------------------------------------------------------------------
// Kernel 2: workgroup = 128-row strip x one column quarter. Wave w holds the
// A strip for rows [r0+16w, r0+16w+16) in 64 VGPRs. B tiles (16x256 halves)
// are double-buffered in LDS, staged by async DMA, shared by all 8 waves.
//
// A layout (16x32 f16): lanes 0-15 row M=lane, K = {0..7, 16..23};
//                       lanes 16-31 row M=lane-16, K = {8..15, 24..31}.
// B layout (32x16 f16): lanes 0-15 col N=lane, K = {0..7, 16..23};
//                       lanes 16-31 col N=lane-16, K = {8..15, 24..31}.
// C layout: VGPR v -> M = v (+8 for lanes 16-31), N = lane&15.
// Because sim = F F^T, B columns are rows of F: all loads are contiguous.
// ---------------------------------------------------------------------------
__global__ __launch_bounds__(256) void ContrastiveLoss_sim(
    const _Float16* __restrict__ F,
    float* __restrict__ accPart, float* __restrict__ posPart) {
  __shared__ __align__(16) _Float16 Bsh[2][TILE][LDSPITCH];

  const int strip = blockIdx.x >> 2;        // 0..63
  const int q     = blockIdx.x & 3;         // 0..3
  const int r0    = strip * STRIP;
  const int c0    = q * QCOLS;

  const int  wave   = threadIdx.x >> 5;
  const int  lane   = threadIdx.x & 31;
  const int  laneLo = lane & 15;
  const bool hi     = lane >= 16;

  // staging coords: 256 threads x 32B cover one 16x256-half tile
  const int srow = threadIdx.x >> 4;        // 0..15
  const int sofs = (threadIdx.x & 15) * 16; // half offset within row

  const uint32_t ldsDst[2] = { lds_byte_addr(&Bsh[0][srow][sofs]),
                               lds_byte_addr(&Bsh[1][srow][sofs]) };

  union V16 { v16h h16; v8h h8[2]; };

  // Load this wave's A strip once: 16 rows x 256 K -> 8 x v16h per lane.
  V16 a[KCH];
  const _Float16* Arow = F + (size_t)(r0 + wave * TILE + laneLo) * DIM;
#pragma unroll
  for (int kk = 0; kk < KCH; ++kk) {
    const int kb = kk * 32 + (hi ? 8 : 0);
    a[kk].h8[0] = *(const v8h*)(Arow + kb);        // K = kb   .. kb+7
    a[kk].h8[1] = *(const v8h*)(Arow + kb + 16);   // K = kb+16.. kb+23
  }

  float acc[8], pos[8];
#pragma unroll
  for (int v = 0; v < 8; ++v) { acc[v] = 0.0f; pos[v] = 0.0f; }

  // Stage tile 0 into buffer 0 (async DMA).
  {
    const _Float16* src = F + (size_t)(c0 + srow) * DIM + sofs;
    async_copy_b128(ldsDst[0],      src);
    async_copy_b128(ldsDst[0] + 16, src + 8);
  }

  for (int t = 0; t < QTILES; ++t) {
    wait_asynccnt0();                        // my async writes for tile t done
    __syncthreads();                         // all writes visible; compute(t-1) done
    const int buf = t & 1;

    if (t + 1 < QTILES) {                    // DMA next tile into the other buffer
      const _Float16* src = F + (size_t)(c0 + (t + 1) * TILE + srow) * DIM + sofs;
      async_copy_b128(ldsDst[buf ^ 1],      src);
      async_copy_b128(ldsDst[buf ^ 1] + 16, src + 8);
    }

    v8f c = {};
#pragma unroll
    for (int kk = 0; kk < KCH; ++kk) {
      const int kb = kk * 32 + (hi ? 8 : 0);
      V16 b;
      b.h8[0] = *(const v8h*)(&Bsh[buf][laneLo][kb]);
      b.h8[1] = *(const v8h*)(&Bsh[buf][laneLo][kb + 16]);
      c = __builtin_amdgcn_wmma_f32_16x16x32_f16(
          /*neg_a=*/false, a[kk].h16, /*neg_b=*/false, b.h16,
          /*c_mod=*/(short)0, c, /*reuse_a=*/false, /*reuse_b=*/false);
    }

    const int n = c0 + t * TILE + laneLo;
#pragma unroll
    for (int v = 0; v < 8; ++v) {
      const int   m = r0 + wave * TILE + v + (hi ? 8 : 0);
      const float s = c[v] * INV_T;
      if (n != m) acc[v] += __expf(s - INV_T);           // diagonal -> exp(-inf)=0
      if (n == ((m + BHALF) & (NROWS - 1))) pos[v] += s; // one hit per row globally
    }
  }

  // Reduce across the 16 lanes sharing each row (xor masks stay within halves).
#pragma unroll
  for (int v = 0; v < 8; ++v) {
#pragma unroll
    for (int msk = 8; msk >= 1; msk >>= 1) {
      acc[v] += __shfl_xor(acc[v], msk, 32);
      pos[v] += __shfl_xor(pos[v], msk, 32);
    }
  }

  // Waves own disjoint rows: write per-row quarter partials directly.
  if (laneLo == 0) {
#pragma unroll
    for (int v = 0; v < 8; ++v) {
      const int row = r0 + wave * TILE + v + (hi ? 8 : 0);
      accPart[q * NROWS + row] = acc[v];
      posPart[q * NROWS + row] = pos[v];
    }
  }
}

// ---------------------------------------------------------------------------
// Kernel 3: combine the 4 column-quarter partials per row, form lse - pos,
// deterministic fixed-order mean over all 8192 rows.
// ---------------------------------------------------------------------------
__global__ __launch_bounds__(256) void ContrastiveLoss_red(
    const float* __restrict__ accPart, const float* __restrict__ posPart,
    float* __restrict__ out) {
  __shared__ float s[256];
  float t = 0.0f;
  for (int row = threadIdx.x; row < NROWS; row += 256) {
    float sa = 0.0f, sp = 0.0f;
#pragma unroll
    for (int qq = 0; qq < NQ; ++qq) {
      sa += accPart[qq * NROWS + row];
      sp += posPart[qq * NROWS + row];
    }
    t += (__logf(sa) + INV_T) - sp;          // lse - pos
  }
  s[threadIdx.x] = t;
  __syncthreads();
  for (int step = 128; step > 0; step >>= 1) {
    if (threadIdx.x < step) s[threadIdx.x] += s[threadIdx.x + step];
    __syncthreads();
  }
  if (threadIdx.x == 0) out[0] = s[0] * (1.0f / NROWS);  // WEIGHT = 1, mean over 2B
}

extern "C" void kernel_launch(void* const* d_in, const int* in_sizes, int n_in,
                              void* d_out, int out_size, void* d_ws, size_t ws_size,
                              hipStream_t stream) {
  (void)in_sizes; (void)n_in; (void)out_size; (void)ws_size;
  const float* f1 = (const float*)d_in[0];
  const float* f2 = (const float*)d_in[1];
  float* out = (float*)d_out;

  _Float16* F = (_Float16*)d_ws;                                   // 4 MB
  char* p = (char*)d_ws + (size_t)NROWS * DIM * sizeof(_Float16);
  float* accPart = (float*)p;                                      // 4*8192 floats
  float* posPart = accPart + (size_t)NQ * NROWS;                   // 4*8192 floats

  ContrastiveLoss_nrm<<<NROWS / 8, 256, 0, stream>>>(f1, f2, F);
  ContrastiveLoss_sim<<<NSTRIPS * NQ, 256, 0, stream>>>(F, accPart, posPart);
  ContrastiveLoss_red<<<1, 256, 0, stream>>>(accPart, posPart, out);
}
